// QuantizedLinear_24799141167655
// MI455X (gfx1250) — compile-verified, matrix operand order checked by reading, not run on hardware
//
#include <hip/hip_runtime.h>

typedef __attribute__((ext_vector_type(16))) _Float16 v16h;
typedef __attribute__((ext_vector_type(8)))  _Float16 v8h;
typedef __attribute__((ext_vector_type(8)))  float    v8f;
typedef __attribute__((ext_vector_type(4)))  float    v4f;
typedef __attribute__((ext_vector_type(4)))  int      v4i;

#define IN_F     4096
#define OUT_F    4096
#define GS       32
#define KGROUPS  (IN_F / GS)   // 128 groups along K per output feature

#define BM 128
#define BN 256
#define BK 32                  // one quant group per K step
#define LDSW 40                // padded row stride in f16 (80B -> conflict-free b128 access)

__global__ __launch_bounds__(256)
void qlinear_wmma_f16_kernel(const float* __restrict__ x,
                             const int*   __restrict__ qweight,
                             const float* __restrict__ scales,
                             float*       __restrict__ out,
                             int M)
{
    __shared__ __attribute__((aligned(16))) _Float16 As[BM * LDSW]; // [m][k] f16, 10 KB
    __shared__ __attribute__((aligned(16))) _Float16 Bs[BN * LDSW]; // [n][k] f16, 20 KB

    const int tid  = threadIdx.x;
    const int lane = tid & 31;
    const int wave = tid >> 5;

    const int m0 = blockIdx.y * BM;
    const int n0 = blockIdx.x * BN;

    // 8 waves: 2 in M x 4 in N; each wave computes 64x64 = 4x4 WMMA tiles
    const int wMi = wave >> 2;     // 0..1
    const int wNi = wave & 3;      // 0..3
    const int lr  = lane & 15;     // row (A) / col (B) within a 16x16 tile
    const int lh  = lane >> 4;     // K-half selector per ISA 16-bit fragment layout

    // Cooperative loads:
    //   A: 2 threads per row, each covers 16 k-floats  (rows 0..127)
    //   B: 1 thread per output feature (group), 16 packed int32 (cols 0..255)
    const int arow  = tid >> 1;          // 0..127
    const int khalf = (tid & 1) * 16;    // 0 or 16
    const int bcol  = tid;               // 0..255

    v8f acc[4][4];
    const v8f vzero = {};
    #pragma unroll
    for (int i = 0; i < 4; ++i)
        #pragma unroll
        for (int j = 0; j < 4; ++j)
            acc[i][j] = vzero;

    // ---- register staging for software pipeline ----
    v4f  a_stage[4];     // 16 floats of x
    v4i  b_stage[4];     // 16 packed int32 = one full group (32 nibbles)
    float s_stage;

    const float* xrow  = x + (size_t)(m0 + arow) * IN_F + khalf;
    const size_t gbase = (size_t)(n0 + bcol) * KGROUPS;

    auto load_stage = [&](int kb) {
        const float* ap = xrow + kb * BK;
        a_stage[0] = *(const v4f*)(ap + 0);
        a_stage[1] = *(const v4f*)(ap + 4);
        a_stage[2] = *(const v4f*)(ap + 8);
        a_stage[3] = *(const v4f*)(ap + 12);
        const size_t gidx = gbase + kb;
        const int* qp = qweight + gidx * 16;
        b_stage[0] = *(const v4i*)(qp + 0);
        b_stage[1] = *(const v4i*)(qp + 4);
        b_stage[2] = *(const v4i*)(qp + 8);
        b_stage[3] = *(const v4i*)(qp + 12);
        s_stage = scales[gidx];
    };

    auto store_stage = [&]() {
        // A: fp32 -> f16
        union { _Float16 h[16]; v8h v[2]; } ua;
        #pragma unroll
        for (int v = 0; v < 4; ++v)
            #pragma unroll
            for (int e = 0; e < 4; ++e)
                ua.h[v * 4 + e] = (_Float16)a_stage[v][e];
        *(v8h*)&As[arow * LDSW + khalf + 0] = ua.v[0];
        *(v8h*)&As[arow * LDSW + khalf + 8] = ua.v[1];

        // B: unpack nibbles, symmetric dequant: scale * (q - 8) -> 32 f16
        union { _Float16 h[32]; v8h v[4]; } ub;
        #pragma unroll
        for (int v = 0; v < 4; ++v)
            #pragma unroll
            for (int e = 0; e < 4; ++e) {
                const int byte = b_stage[v][e];
                const int idx  = (v * 4 + e) * 2;
                ub.h[idx]     = (_Float16)(s_stage * (float)((byte & 15) - 8));
                ub.h[idx + 1] = (_Float16)(s_stage * (float)(((byte >> 4) & 15) - 8));
            }
        _Float16* bp = &Bs[bcol * LDSW];
        *(v8h*)(bp + 0)  = ub.v[0];
        *(v8h*)(bp + 8)  = ub.v[1];
        *(v8h*)(bp + 16) = ub.v[2];
        *(v8h*)(bp + 24) = ub.v[3];
    };

    // Fragment loads per ISA 7.12.2 (16-bit A 16x32: lanes 0-15 K=0..7,16..23;
    // lanes 16-31 K=8..15,24..31). B stored K-major per column -> same addressing.
    auto load_a_frag = [&](int i) -> v16h {
        const _Float16* p = &As[(wMi * 64 + i * 16 + lr) * LDSW];
        union { v16h f; v8h v[2]; } u;
        u.v[0] = *(const v8h*)(p + lh * 8);
        u.v[1] = *(const v8h*)(p + 16 + lh * 8);
        return u.f;
    };
    auto load_b_frag = [&](int j) -> v16h {
        const _Float16* p = &Bs[(wNi * 64 + j * 16 + lr) * LDSW];
        union { v16h f; v8h v[2]; } u;
        u.v[0] = *(const v8h*)(p + lh * 8);
        u.v[1] = *(const v8h*)(p + 16 + lh * 8);
        return u.f;
    };

    load_stage(0);

    for (int kb = 0; kb < KGROUPS; ++kb) {
        __syncthreads();          // previous tile fully consumed
        store_stage();
        __syncthreads();          // tile visible to all waves

        if (kb + 1 < KGROUPS)
            load_stage(kb + 1);   // overlap HBM latency with WMMA below
        if (kb + 2 < KGROUPS)
            __builtin_prefetch(xrow + (kb + 2) * BK, 0, 1);  // global_prefetch_b8

        v16h afr[4];
        #pragma unroll
        for (int i = 0; i < 4; ++i) afr[i] = load_a_frag(i);

        // Stream B fragments to keep peak VGPR pressure < 256
        #pragma unroll
        for (int j = 0; j < 4; ++j) {
            const v16h bfr = load_b_frag(j);
            #pragma unroll
            for (int i = 0; i < 4; ++i)
                acc[i][j] = __builtin_amdgcn_wmma_f32_16x16x32_f16(
                    /*neg_a=*/false, afr[i], /*neg_b=*/false, bfr,
                    /*c_mod=*/(short)0, acc[i][j],
                    /*reuse_a=*/false, /*reuse_b=*/false);
        }
    }

    // Epilogue: C layout (lanes 0-15: N=lane, M=r; lanes 16-31: N=lane-16, M=8+r)
    #pragma unroll
    for (int i = 0; i < 4; ++i) {
        const int gm = m0 + wMi * 64 + i * 16 + lh * 8;
        #pragma unroll
        for (int j = 0; j < 4; ++j) {
            const int gn = n0 + wNi * 64 + j * 16 + lr;
            #pragma unroll
            for (int r = 0; r < 8; ++r)
                out[(size_t)(gm + r) * OUT_F + gn] = acc[i][j][r];
        }
    }
    (void)M;
}

extern "C" void kernel_launch(void* const* d_in, const int* in_sizes, int n_in,
                              void* d_out, int out_size, void* d_ws, size_t ws_size,
                              hipStream_t stream) {
    const float* x       = (const float*)d_in[0];
    const int*   qweight = (const int*)d_in[1];
    const float* scales  = (const float*)d_in[2];
    float*       out     = (float*)d_out;

    const int M = in_sizes[0] / IN_F;          // 4*2048 = 8192
    dim3 grid(OUT_F / BN, M / BM);             // (16, 64)
    dim3 block(256);
    qlinear_wmma_f16_kernel<<<grid, block, 0, stream>>>(x, qweight, scales, out, M);
    (void)n_in; (void)out_size; (void)d_ws; (void)ws_size;
}
